// MTM_22239340659316
// MI455X (gfx1250) — matrix operand order repaired and checked by reading, not compiled, not run on hardware
//
#include <hip/hip_runtime.h>
#include <math.h>
#include <string.h>

typedef __attribute__((ext_vector_type(16))) __bf16 v16bf;
typedef __attribute__((ext_vector_type(8)))  float  v8f;
typedef __attribute__((ext_vector_type(4)))  unsigned u32x4;

#define NBLK 100
#define NTHR 256

struct Sched { unsigned s[64]; };

union ABv { u32x4 q[2]; v16bf v; };

__device__ __forceinline__ unsigned short f2bf(float f){
  unsigned u = __float_as_uint(f);
  u = (u + 0x7FFFu + ((u >> 16) & 1u)) >> 16;
  return (unsigned short)u;
}

// ---------------- device-wide sense-reversing barrier ----------------
__device__ __forceinline__ void gsync(unsigned* cnt, unsigned* gen, unsigned nb){
  __syncthreads();
  if(threadIdx.x == 0){
    __threadfence();
    unsigned g = __hip_atomic_load(gen, __ATOMIC_ACQUIRE, __HIP_MEMORY_SCOPE_AGENT);
    unsigned a = __hip_atomic_fetch_add(cnt, 1u, __ATOMIC_ACQ_REL, __HIP_MEMORY_SCOPE_AGENT);
    if(a == nb - 1u){
      __hip_atomic_store(cnt, 0u, __ATOMIC_RELAXED, __HIP_MEMORY_SCOPE_AGENT);
      __hip_atomic_fetch_add(gen, 1u, __ATOMIC_ACQ_REL, __HIP_MEMORY_SCOPE_AGENT);
    } else {
      while(__hip_atomic_load(gen, __ATOMIC_ACQUIRE, __HIP_MEMORY_SCOPE_AGENT) == g)
        __builtin_amdgcn_s_sleep(2);
    }
  }
  __syncthreads();
}

// ------------- stage input vector (fp32 gather -> bf16 LDS) ----------
__device__ __forceinline__ void stagev(unsigned short* vldsp,
    const float* const* segs, const int* lens, int nseg, int K)
{
  for(int i = threadIdx.x; i < K; i += NTHR){
    int j = i, s = 0;
    while(s < nseg - 1 && j >= lens[s]){ j -= lens[s]; s++; }
    vldsp[i] = f2bf(segs[s][j]);
  }
  __syncthreads();
}

__device__ __forceinline__ void copy768(float* dst, const float* src){
  for(int i = threadIdx.x; i < 768; i += NTHR) dst[i] = src[i];
}

// ---- 16-row mat-vec block: y[16] = act(W[16xK] @ v[K] + b), WMMA bf16 ----
// wave w handles K-chunk [w*K/8, (w+1)*K/8), 32 per WMMA.
// A layout (16-bit 16x32): lane m<16 -> row m, K 0..7 / 16..23; lanes 16..31 -> K 8..15 / 24..31.
// B: v broadcast across all 16 columns -> every column of D equals y-partial.
__device__ __forceinline__ void matvec16(
    const unsigned short* __restrict__ Wrow, int K,
    const float* __restrict__ bias16, bool sigm,
    float* __restrict__ out16, float* __restrict__ out16b,
    const unsigned short* vldsp, float* red)
{
  const int tid  = threadIdx.x;
  const int lane = tid & 31;
  const int w    = tid >> 5;
  const int m    = lane & 15;
  const int hi   = lane >> 4;          // 0 or 1
  const int kpw  = K >> 3;             // K per wave (multiple of 32 for all our K)
  const int k0   = w * kpw;
  const unsigned short* rp = Wrow + (size_t)m * (size_t)K + (hi << 3);
  v8f c = {0.f,0.f,0.f,0.f,0.f,0.f,0.f,0.f};
  for(int k = k0; k < k0 + kpw; k += 32){
    ABv a, bm;
    const unsigned short* ap = rp + k;
    __builtin_prefetch((const void*)(ap + 128), 0, 1);   // stream weights ahead
    a.q[0] = *(const u32x4*)(ap);
    a.q[1] = *(const u32x4*)(ap + 16);
    const unsigned short* bp = vldsp + k + (hi << 3);
    bm.q[0] = *(const u32x4*)(bp);
    bm.q[1] = *(const u32x4*)(bp + 16);
    c = __builtin_amdgcn_wmma_f32_16x16x32_bf16(false, a.v, false, bm.v,
                                                (short)0, c, false, false);
  }
  // C layout: VGPR j -> M=j (lanes 0..15) / M=j+8 (lanes 16..31); all columns equal.
  if(lane == 0){
    #pragma unroll
    for(int j = 0; j < 8; j++) red[w*16 + j] = c[j];
  } else if(lane == 16){
    #pragma unroll
    for(int j = 0; j < 8; j++) red[w*16 + 8 + j] = c[j];
  }
  __syncthreads();
  if(tid < 16){
    float s = 0.f;
    #pragma unroll
    for(int q = 0; q < 8; q++) s += red[q*16 + tid];
    s += bias16[tid];
    if(sigm) s = 1.f / (1.f + __expf(-s));
    out16[tid] = s;
    if(out16b) out16b[tid] = s;
  }
  __syncthreads();
}

// ---------------- helper kernels ----------------
__global__ __launch_bounds__(256) void mtm_zero_f(float* p, int n){
  int i = blockIdx.x * 256 + threadIdx.x;
  for(; i < n; i += gridDim.x * 256) p[i] = 0.f;
}
__global__ __launch_bounds__(256) void mtm_cvt_bf16(const float* __restrict__ src,
                                                    unsigned short* __restrict__ dst, int n){
  int i = blockIdx.x * 256 + threadIdx.x;
  for(; i < n; i += gridDim.x * 256) dst[i] = f2bf(src[i]);
}

// ---------------- persistent scan kernel ----------------
__global__ __launch_bounds__(NTHR, 1) void mtm_main(
  const float* __restrict__ x,
  const float* __restrict__ o0b, const float* __restrict__ lb,
  const float* __restrict__ rbv, const float* __restrict__ ob,
  const float* __restrict__ outb,
  const unsigned short* __restrict__ w_o0, const unsigned short* __restrict__ w_l,
  const unsigned short* __restrict__ w_r,  const unsigned short* __restrict__ w_o,
  const unsigned short* __restrict__ w_out,
  float* hid1, float* hidden, float* tmpb, float* ubuf, float* sbuf,
  unsigned* cnt, unsigned* gen, float* outp, Sched sc, unsigned nblocks)
{
  __shared__ __align__(16) unsigned short vlds[5632];
  __shared__ float red[128];
  const int b = blockIdx.x;

  for(int t = 0; t < 64; t++){
    const unsigned bits = sc.s[t];
    const int idv   = bits & 127;
    const int hstar = (bits >> 19) & 7;
    const int side  = (bits >> 22) & 1;
    const float* xt = x + t * 256;
    const int ping  = t & 1;
    float* h1prev = hid1 + ping * 768;
    float* h1cur  = hid1 + (1 - ping) * 768;

    // ---- round: other0  (hidden[1] = W @ [x; h1], no sigmoid) ----
    if(b < 48){
      const float* segs[2] = { xt, h1prev }; int lens[2] = { 256, 768 };
      stagev(vlds, segs, lens, 2, 1024);
      matvec16(w_o0 + (size_t)b * 16 * 1024, 1024, o0b + b * 16,
               false, h1cur + b * 16, nullptr, vlds, red);
    }
    gsync(cnt, gen, nblocks);

    int now = 1, pls = 0, prs = 0;
    for(int h = 1; h <= 6; h++){
      const int ls = 2 * now, rs = ls + 1;
      const int gl = (bits >> (7 + h - 1)) & 1;
      const int cd = (bits >> (13 + h - 1)) & 1;
      const float* hnow;
      if(h == 1) hnow = h1cur;
      else {
        int glp = (bits >> (7 + h - 2)) & 1;
        hnow = tmpb + (size_t)((h - 2) * 2 + (glp ? 0 : 1)) * 768;
      }

      // ---- round A: u = sigmoid(O @ [h_now; child]) (only when cond) ----
      if(cd){
        if(b < 96){
          const int j = b / 48, rbk = b % 48;
          const float* segs[2] = { hnow, hidden + (size_t)(j ? rs : ls) * 768 };
          int lens[2] = { 768, 768 };
          stagev(vlds, segs, lens, 2, 1536);
          matvec16(w_o + ((size_t)((h - 1) * 2 + j) * 768 + (size_t)rbk * 16) * 1536,
                   1536, ob + ((h - 1) * 2 + j) * 768 + rbk * 16,
                   true, ubuf + j * 768 + rbk * 16, nullptr, vlds, red);
        } else if(h >= 2 && (b == 96 || b == 97)){
          int j = b - 96;   // commit previous height's children
          copy768(hidden + (size_t)(j ? prs : pls) * 768, tmpb + (size_t)((h - 2) * 2 + j) * 768);
        }
        gsync(cnt, gen, nblocks);
      }

      // ---- round B: new child = sigmoid(W @ [x; h_now; u|child]) ----
      const unsigned short* WLR = (gl ? w_l : w_r);
      const float* bLR = (gl ? lb : rbv) + ((h - 1) * 2) * 768;
      if(b < 96){
        const int j = b / 48, rbk = b % 48;
        const float* third = cd ? (ubuf + j * 768) : (hidden + (size_t)(j ? rs : ls) * 768);
        const float* segs[3] = { xt, hnow, third }; int lens[3] = { 256, 768, 768 };
        stagev(vlds, segs, lens, 3, 1792);
        float* dup = (hstar == h && side == j) ? (sbuf + (h - 1) * 768 + rbk * 16) : nullptr;
        matvec16(WLR + ((size_t)((h - 1) * 2 + j) * 768 + (size_t)rbk * 16) * 1792,
                 1792, bLR + j * 768 + rbk * 16,
                 true, tmpb + (size_t)((h - 1) * 2 + j) * 768 + rbk * 16, dup, vlds, red);
      } else if(!cd && h >= 2 && (b == 96 || b == 97)){
        int j = b - 96;
        copy768(hidden + (size_t)(j ? prs : pls) * 768, tmpb + (size_t)((h - 2) * 2 + j) * 768);
      } else if(b == 98 && hstar != h){
        // snapshot s[h-1] = hidden[idv] (stable source per host schedule)
        const float* src;
        if(hstar == 7) src = h1cur;
        else if(hstar >= 1 && hstar < h) src = tmpb + (size_t)((hstar - 1) * 2 + side) * 768;
        else src = hidden + (size_t)idv * 768;
        copy768(sbuf + (h - 1) * 768, src);
      }
      gsync(cnt, gen, nblocks);

      pls = ls; prs = rs; now = gl ? ls : rs;
    }

    // ---- round: output  y = sigmoid(Wout @ [x; s1..s6; s6]) ----
    if(b < 16){
      const float* segs[8] = { xt, sbuf, sbuf + 768, sbuf + 2*768, sbuf + 3*768,
                               sbuf + 4*768, sbuf + 5*768, sbuf + 5*768 };
      int lens[8] = { 256, 768, 768, 768, 768, 768, 768, 768 };
      stagev(vlds, segs, lens, 8, 5632);
      matvec16(w_out + (size_t)b * 16 * 5632, 5632, outb + b * 16,
               true, outp + t * 256 + b * 16, nullptr, vlds, red);
    } else if(b == 16 || b == 17){
      int j = b - 16;   // commit height-6 children
      copy768(hidden + (size_t)(j ? prs : pls) * 768, tmpb + (size_t)(5 * 2 + j) * 768);
    }
    gsync(cnt, gen, nblocks);
  }
}

// ---------------- host ----------------
extern "C" void kernel_launch(void* const* d_in, const int* in_sizes, int n_in,
                              void* d_out, int out_size, void* d_ws, size_t ws_size,
                              hipStream_t stream)
{
  const float* x    = (const float*)d_in[0];
  const float* o0w  = (const float*)d_in[1];
  const float* o0b  = (const float*)d_in[2];
  const float* lw   = (const float*)d_in[3];
  const float* lb   = (const float*)d_in[4];
  const float* rw   = (const float*)d_in[5];
  const float* rb   = (const float*)d_in[6];
  const float* ow   = (const float*)d_in[7];
  const float* ob   = (const float*)d_in[8];
  const float* outw = (const float*)d_in[9];
  const float* outb = (const float*)d_in[10];

  const size_t N_O0 = 768ull * 1024ull;
  const size_t N_LR = 6ull * 2 * 768 * 1792;
  const size_t N_OT = 6ull * 2 * 768 * 1536;
  const size_t N_OW = 256ull * 5632ull;

  float* f = (float*)d_ws;
  unsigned* cnt = (unsigned*)d_ws;
  unsigned* gen = (unsigned*)((char*)d_ws + 128);
  float* hid1   = f + 64;              // 2 x 768  (double-buffered node 1)
  float* hidden = hid1 + 2 * 768;      // 129 x 768
  float* tmpb   = hidden + 129 * 768;  // 6 heights x 2 children x 768
  float* ubuf   = tmpb + 12 * 768;     // 2 x 768
  float* sbuf   = ubuf + 2 * 768;      // 6 x 768 snapshots
  unsigned short* wb = (unsigned short*)(sbuf + 6 * 768);
  unsigned short* w_o0 = wb;
  unsigned short* w_l  = w_o0 + N_O0;
  unsigned short* w_r  = w_l + N_LR;
  unsigned short* w_o  = w_r + N_LR;
  unsigned short* w_ow = w_o + N_OT;
  size_t need = (size_t)((char*)(w_ow + N_OW) - (char*)d_ws);
  if(ws_size < need) return;

  // ---- host-side exact simulation of the tree control flow ----
  Sched sc;
  int change[130]; for(int i = 0; i < 130; i++) change[i] = 0;
  int idv = 1;
  for(int t = 0; t < 64; t++){
    unsigned bits = 0;
    int now = 1, L = 1, R = 64;
    int lsA[7], rsA[7], glA[7];
    for(int h = 1; h <= 6; h++){
      int ls = 2 * now, rs = ls + 1, M = (L + R) / 2;
      int cd = (change[now] == 0);
      change[idv] = 0;
      int gl = (idv <= M);
      bits |= ((unsigned)gl) << (7 + h - 1);
      bits |= ((unsigned)cd) << (13 + h - 1);
      lsA[h] = ls; rsA[h] = rs; glA[h] = gl;
      if(gl){ now = ls; R = M; } else { now = rs; L = M + 1; }
    }
    for(int h = 6; h >= 1; h--) change[glA[h] ? rsA[h] : lsA[h]] = 1;
    int hstar = 0, side = 0;
    if(idv == 1) hstar = 7;
    else {
      int p = idv >> 1, nn = 1;
      for(int h = 1; h <= 6; h++){
        if(nn == p){ hstar = h; side = idv & 1; break; }
        nn = glA[h] ? lsA[h] : rsA[h];
      }
    }
    bits |= (unsigned)(idv & 127);
    bits |= ((unsigned)hstar) << 19;
    bits |= ((unsigned)side) << 22;
    sc.s[t] = bits;
    idv++; if(idv > 64) idv -= 64;
  }

  // zero state + barrier counters (graph-replay safe: runs each call)
  int nzero = 64 + 151 * 768;
  hipLaunchKernelGGL(mtm_zero_f, dim3(128), dim3(256), 0, stream, f, nzero);

  // one-shot fp32 -> bf16 weight conversion into workspace
  hipLaunchKernelGGL(mtm_cvt_bf16, dim3(2048), dim3(256), 0, stream, o0w,  w_o0, (int)N_O0);
  hipLaunchKernelGGL(mtm_cvt_bf16, dim3(4096), dim3(256), 0, stream, lw,   w_l,  (int)N_LR);
  hipLaunchKernelGGL(mtm_cvt_bf16, dim3(4096), dim3(256), 0, stream, rw,   w_r,  (int)N_LR);
  hipLaunchKernelGGL(mtm_cvt_bf16, dim3(4096), dim3(256), 0, stream, ow,   w_o,  (int)N_OT);
  hipLaunchKernelGGL(mtm_cvt_bf16, dim3(1024), dim3(256), 0, stream, outw, w_ow, (int)N_OW);

  // persistent cooperative scan
  hipLaunchKernelGGL(mtm_main, dim3(NBLK), dim3(NTHR), 0, stream,
      x, o0b, lb, rb, ob, outb,
      w_o0, w_l, w_r, w_o, w_ow,
      hid1, hidden, tmpb, ubuf, sbuf,
      cnt, gen, (float*)d_out, sc, (unsigned)NBLK);
  (void)in_sizes; (void)n_in; (void)out_size;
}
